// Multihead_GaAtN_70506183131635
// MI455X (gfx1250) — matrix operand order repaired and checked by reading, not compiled, not run on hardware
//
#include <hip/hip_runtime.h>
#include <hip/hip_bf16.h>
#include <stdint.h>

// ---- problem constants (match reference) ----
#define NN   50000   // nodes
#define DEG  32      // neighbors per node
#define DIMF 128     // feature dim
#define NH   4       // heads
// N is divisible by 16: 50000 = 3125 * 16

typedef __attribute__((ext_vector_type(16))) __bf16 v16bf;
typedef __attribute__((ext_vector_type(8)))  __bf16 bf16x8;
typedef __attribute__((ext_vector_type(4)))  __bf16 bf16x4;
typedef __attribute__((ext_vector_type(8)))  float  v8f;

__device__ __forceinline__ float  bf2f(__bf16 x) { return (float)x; }
__device__ __forceinline__ __bf16 f2bf(float x)  { return (__bf16)x; }

__device__ __forceinline__ float wave_sum(float v) {
#pragma unroll
  for (int o = 16; o > 0; o >>= 1) v += __shfl_xor(v, o, 32);
  return v;
}
__device__ __forceinline__ float wave_max(float v) {
#pragma unroll
  for (int o = 16; o > 0; o >>= 1) v = fmaxf(v, __shfl_xor(v, o, 32));
  return v;
}

// ---------------- conversion / transpose prep ----------------
__global__ __launch_bounds__(256) void cvt_f32_to_bf16(const float* __restrict__ in,
                                                       __bf16* __restrict__ out, int n) {
  int i = blockIdx.x * 256 + threadIdx.x;
  if (i < n) out[i] = f2bf(in[i]);
}

// Wt[(k*128+j)*128 + i] = W[k][i][j]   (column-major per output column, bf16)
__global__ __launch_bounds__(256) void cvt_W(const float* __restrict__ W,
                                             __bf16* __restrict__ Wt) {
  int t = blockIdx.x * 256 + threadIdx.x;      // 4*128*128 = 65536
  int k = t >> 14, rem = t & 16383;
  int i = rem >> 7, j = rem & 127;
  Wt[((((size_t)k << 7) | j) << 7) | i] = f2bf(W[t]);
}

// W0t[j*512 + i] = W0[i*128 + j]
__global__ __launch_bounds__(256) void cvt_W0(const float* __restrict__ W0,
                                              __bf16* __restrict__ W0t) {
  int t = blockIdx.x * 256 + threadIdx.x;      // 512*128 = 65536
  int i = t >> 7, j = t & 127;
  W0t[((size_t)j << 9) | i] = f2bf(W0[t]);
}

// ---------------- WMMA bf16 GEMM: C[M x (Ntiles*16)] = A[M x K] * Bt^T ----------------
// A row-major [row][K] contiguous; Bt is B transposed: Bt[col][K] contiguous.
// One wave computes NB adjacent 16x16 C tiles (N-blocking): each A fragment
// loaded once per K-step feeds NB chained v_wmma_f32_16x16x32_bf16, cutting
// A-operand L2 traffic by NB x.
template <int KDIM, int NB, bool OUT_BF16>
__global__ __launch_bounds__(256) void gemm_bf16_wmma(const __bf16* __restrict__ A,
                                                      const __bf16* __restrict__ Bt,
                                                      void* __restrict__ Cptr,
                                                      int Mtiles, int Ntiles) {
  int ngroups = Ntiles / NB;
  int wid = (blockIdx.x * 256 + (int)threadIdx.x) >> 5;   // task id (wave-uniform)
  if (wid >= Mtiles * ngroups) return;
  int tm  = wid / ngroups;
  int tn0 = (wid % ngroups) * NB;
  int lane = threadIdx.x & 31;
  int hl = lane >> 4;          // lane half (0 / 1)
  int r  = lane & 15;          // A row within tile / B column within tile

  const __bf16* arow  = A  + (size_t)(tm * 16 + r) * KDIM;
  const __bf16* bcol0 = Bt + (size_t)(tn0 * 16 + r) * KDIM;

  v8f c[NB];
#pragma unroll
  for (int b = 0; b < NB; ++b) c[b] = v8f{};

#pragma unroll
  for (int ks = 0; ks < KDIM / 32; ++ks) {
    int kb = ks * 32;
    // A fragment (16x32): e<8 -> K = kb + hl*8 + e ; e>=8 -> K = kb + 16 + hl*8 + (e-8)
    bf16x8 a_lo = *(const bf16x8*)(arow + kb + hl * 8);
    bf16x8 a_hi = *(const bf16x8*)(arow + kb + 16 + hl * 8);
    v16bf av = __builtin_shufflevector(a_lo, a_hi, 0,1,2,3,4,5,6,7,8,9,10,11,12,13,14,15);
#pragma unroll
    for (int b = 0; b < NB; ++b) {
      // B fragment (32x16): e<8 -> K = kb + hl*8 + e ; e>=8 -> K = kb + 16 + hl*8 + (e-8), column = (tn0+b)*16 + r
      const __bf16* bc = bcol0 + (size_t)b * 16 * KDIM;
      bf16x8 b_lo = *(const bf16x8*)(bc + kb + hl * 8);
      bf16x8 b_hi = *(const bf16x8*)(bc + kb + 16 + hl * 8);
      v16bf bv = __builtin_shufflevector(b_lo, b_hi, 0,1,2,3,4,5,6,7,8,9,10,11,12,13,14,15);
      c[b] = __builtin_amdgcn_wmma_f32_16x16x32_bf16(false, av, false, bv,
                                                     (short)0, c[b], false, false);
    }
  }
  // C layout: element e -> M = hl*8 + e, N = r
  int Ncols = Ntiles * 16;
#pragma unroll
  for (int b = 0; b < NB; ++b) {
    int col = (tn0 + b) * 16 + r;
    if (OUT_BF16) {
      __bf16* C = (__bf16*)Cptr;
#pragma unroll
      for (int e = 0; e < 8; ++e)
        C[(size_t)(tm * 16 + hl * 8 + e) * Ncols + col] = f2bf(c[b][e]);
    } else {
      float* C = (float*)Cptr;
#pragma unroll
      for (int e = 0; e < 8; ++e)
        C[(size_t)(tm * 16 + hl * 8 + e) * Ncols + col] = c[b][e];
    }
  }
}

// ---------------- per-node scalars: s1, s2 (from WH), gate softmax (from h) ----------------
__global__ __launch_bounds__(256) void node_scalars(const __bf16* __restrict__ WH,
                                                    const float* __restrict__ h,
                                                    const float* __restrict__ a1,
                                                    const float* __restrict__ a2,
                                                    const float* __restrict__ Wg,
                                                    float* __restrict__ s1,
                                                    float* __restrict__ s2,
                                                    float* __restrict__ g) {
  int n = (blockIdx.x * 256 + (int)threadIdx.x) >> 5;
  int lane = threadIdx.x & 31;
  if (n >= NN) return;

  float gl[NH] = {0.f, 0.f, 0.f, 0.f};
  for (int i = lane; i < DIMF; i += 32) {
    float hv = h[(size_t)n * DIMF + i];
#pragma unroll
    for (int k = 0; k < NH; ++k) gl[k] += hv * Wg[i * NH + k];
  }
  float s1v[NH], s2v[NH];
#pragma unroll
  for (int k = 0; k < NH; ++k) {
    float p1 = 0.f, p2 = 0.f;
    for (int j = lane; j < DIMF; j += 32) {
      float w = bf2f(WH[(size_t)n * (NH * DIMF) + k * DIMF + j]);
      p1 += w * a1[k * DIMF + j];
      p2 += w * a2[k * DIMF + j];
    }
    s1v[k] = wave_sum(p1);
    s2v[k] = wave_sum(p2);
  }
#pragma unroll
  for (int k = 0; k < NH; ++k) gl[k] = wave_sum(gl[k]);

  if (lane == 0) {
    float mx = fmaxf(fmaxf(gl[0], gl[1]), fmaxf(gl[2], gl[3]));
    float ex[NH], ss = 0.f;
#pragma unroll
    for (int k = 0; k < NH; ++k) { ex[k] = __expf(gl[k] - mx); ss += ex[k]; }
    float inv = 1.f / ss;
#pragma unroll
    for (int k = 0; k < NH; ++k) {
      g[n * NH + k]  = ex[k] * inv;
      s1[n * NH + k] = s1v[k];
      s2[n * NH + k] = s2v[k];
    }
  }
}

// ---------------- attention softmax + gather-aggregate ----------------
// one wave per (node, head); lane d owns neighbor d; 128 outputs -> 4 per lane
__global__ __launch_bounds__(256) void attn_agg(const __bf16* __restrict__ WH,
                                                const int* __restrict__ idx,
                                                const float* __restrict__ s1,
                                                const float* __restrict__ s2,
                                                const float* __restrict__ g,
                                                __bf16* __restrict__ OUTh) {
  int task = (blockIdx.x * 256 + (int)threadIdx.x) >> 5;   // n*NH + k
  int lane = threadIdx.x & 31;
  if (task >= NN * NH) return;
  int n = task >> 2, k = task & 3;

  int j = idx[(size_t)n * DEG + lane];
  float e = s1[n * NH + k] + s2[j * NH + k];
  e = (e > 0.f) ? e : 0.01f * e;                 // leaky_relu, slope 0.01

  float m  = wave_max(e);
  float w  = __expf(e - m);
  float sw = wave_sum(w);
  float att = w / sw;

  float acc0 = 0.f, acc1 = 0.f, acc2 = 0.f, acc3 = 0.f;
#pragma unroll 4
  for (int d = 0; d < DEG; ++d) {
    float ad = __shfl(att, d, 32);
    int   jd = __shfl(j,   d, 32);
    bf16x4 v = *(const bf16x4*)(WH + (size_t)jd * (NH * DIMF) + k * DIMF + lane * 4);
    acc0 += ad * bf2f(v[0]);
    acc1 += ad * bf2f(v[1]);
    acc2 += ad * bf2f(v[2]);
    acc3 += ad * bf2f(v[3]);
  }
  float gk = g[n * NH + k];
  bf16x4 o;
  o[0] = f2bf(gk * fmaxf(acc0, 0.f));
  o[1] = f2bf(gk * fmaxf(acc1, 0.f));
  o[2] = f2bf(gk * fmaxf(acc2, 0.f));
  o[3] = f2bf(gk * fmaxf(acc3, 0.f));
  *(bf16x4*)(OUTh + (size_t)n * (NH * DIMF) + k * DIMF + lane * 4) = o;
}

// ---------------- launcher ----------------
extern "C" void kernel_launch(void* const* d_in, const int* in_sizes, int n_in,
                              void* d_out, int out_size, void* d_ws, size_t ws_size,
                              hipStream_t stream) {
  const float* h   = (const float*)d_in[0];
  const int*   idx = (const int*)  d_in[1];
  const float* W   = (const float*)d_in[2];
  const float* a1  = (const float*)d_in[3];
  const float* a2  = (const float*)d_in[4];
  const float* Wg  = (const float*)d_in[5];
  const float* W0  = (const float*)d_in[6];

  char* ws = (char*)d_ws;
  size_t off = 0;
  auto alloc = [&](size_t bytes) -> void* {
    void* p = ws + off;
    off = (off + bytes + 255) & ~(size_t)255;
    return p;
  };
  __bf16* h_bf = (__bf16*)alloc((size_t)NN * DIMF * 2);          // 12.8 MB
  __bf16* Wt   = (__bf16*)alloc((size_t)512 * 128 * 2);          // 128 KB
  __bf16* W0t  = (__bf16*)alloc((size_t)128 * 512 * 2);          // 128 KB
  __bf16* WH   = (__bf16*)alloc((size_t)NN * 512 * 2);           // 51.2 MB
  __bf16* OUTh = (__bf16*)alloc((size_t)NN * 512 * 2);           // 51.2 MB
  float*  s1   = (float*) alloc((size_t)NN * NH * 4);
  float*  s2   = (float*) alloc((size_t)NN * NH * 4);
  float*  g    = (float*) alloc((size_t)NN * NH * 4);

  // prep: bf16 conversions / transposes
  cvt_f32_to_bf16<<<(NN * DIMF + 255) / 256, 256, 0, stream>>>(h, h_bf, NN * DIMF);
  cvt_W <<<65536 / 256, 256, 0, stream>>>(W,  Wt);
  cvt_W0<<<65536 / 256, 256, 0, stream>>>(W0, W0t);

  // WH[N,512] = h[N,128] @ concat_k W[k]
  // WMMA bf16, 3125 row-tiles x (32/4) N-groups = 25000 waves -> 3125 blocks (exact)
  gemm_bf16_wmma<128, 4, true><<<(3125 * 8 + 7) / 8, 256, 0, stream>>>(
      h_bf, Wt, (void*)WH, 3125, 32);

  // s1/s2 per (node, head) + gate softmax: one wave per node
  node_scalars<<<(NN * 32) / 256, 256, 0, stream>>>(WH, h, a1, a2, Wg, s1, s2, g);

  // per-(node,head) softmax over 32 neighbors + gated gather-aggregate
  attn_agg<<<(NN * NH * 32) / 256, 256, 0, stream>>>(WH, idx, s1, s2, g, OUTh);

  // out[N,128] = OUTh[N,512] @ W0[512,128]
  // WMMA bf16, 3125 row-tiles x (8/4) N-groups = 6250 waves -> 782 blocks
  gemm_bf16_wmma<512, 4, false><<<(3125 * 2 + 7) / 8, 256, 0, stream>>>(
      OUTh, W0t, d_out, 3125, 8);

  (void)in_sizes; (void)n_in; (void)out_size; (void)ws_size;
}
